// ImprovedGenerator_5463198400822
// MI455X (gfx1250) — matrix operand order repaired and checked by reading, compiled, not run on hardware
//
#include <hip/hip_runtime.h>

// ---------------------------------------------------------------------------
// Types for CDNA5 WMMA (wave32, v_wmma_f32_16x16x32_bf16)
// ---------------------------------------------------------------------------
typedef __attribute__((ext_vector_type(16))) __bf16 v16bf;
typedef __attribute__((ext_vector_type(8)))  __bf16 v8bf;
typedef __attribute__((ext_vector_type(8)))  float  v8f;

static __device__ __forceinline__ __bf16 f2bf(float f) {
  union { float f; unsigned u; } a; a.f = f;
  unsigned r = a.u + 0x7FFFu + ((a.u >> 16) & 1u);
  unsigned short h = (unsigned short)(r >> 16);
  union { unsigned short s; __bf16 b; } o; o.s = h; return o.b;
}
static __device__ __forceinline__ float bf2f(__bf16 b) {
  union { unsigned short s; __bf16 b; } i; i.b = b;
  union { float f; unsigned u; } o; o.u = ((unsigned)i.s) << 16; return o.f;
}
static __device__ __forceinline__ v8f vzero8() {
  v8f z;
#pragma unroll
  for (int i = 0; i < 8; ++i) z[i] = 0.f;
  return z;
}
static __device__ __forceinline__ float sigm(float x) {
  return 1.f / (1.f + __expf(-x));
}

// CDNA5 async DMA: global -> LDS, 16B per lane, tracked by ASYNCcnt.
static __device__ __forceinline__ void async_copy_b128(unsigned lds_off, const void* gptr) {
  asm volatile("global_load_async_to_lds_b128 %0, %1, off"
               :: "v"(lds_off), "v"(gptr) : "memory");
}
static __device__ __forceinline__ void wait_async0() {
  asm volatile("s_wait_asynccnt 0x0" ::: "memory");
}

union FragU { v16bf v; v8bf h[2]; };

// A-fragment: 16x32 bf16, lane<16: row=lane, K halves {k0..k0+7, k0+16..k0+23};
// lane>=16: row=lane-16, K halves {k0+8..k0+15, k0+24..k0+31}.
static __device__ __forceinline__ v16bf load_frag_a(const __bf16* base, int ld,
                                                    int row0, int k0, int lane) {
  int m = row0 + (lane & 15);
  const __bf16* p = base + (size_t)m * ld + k0 + ((lane >> 4) ? 8 : 0);
  FragU u;
  u.h[0] = *(const v8bf*)p;
  u.h[1] = *(const v8bf*)(p + 16);
  return u.v;
}
// B-fragment: 32x16 (KxN) bf16, lane<16: col=lane, K halves {k0..k0+7, k0+16..k0+23};
// lane>=16: col=lane-16, K halves {k0+8..k0+15, k0+24..k0+31}.  Row n of W == column n of W^T.
static __device__ __forceinline__ v16bf load_frag_b(const __bf16* base, int ld,
                                                    int col0, int k0, int lane) {
  int n = col0 + (lane & 15);
  const __bf16* p = base + (size_t)n * ld + k0 + ((lane >> 4) ? 8 : 0);
  FragU u;
  u.h[0] = *(const v8bf*)p;
  u.h[1] = *(const v8bf*)(p + 16);
  return u.v;
}

// ---------------------------------------------------------------------------
// Weight prep: Wcat = [W_ih | W_hh] bf16, combined bias; bf16 Wq/Wk/Wv copies
// ---------------------------------------------------------------------------
__global__ void __launch_bounds__(256)
prep_kernel(const float* Wih0, const float* Whh0, const float* bih0, const float* bhh0,
            const float* Wih1, const float* Whh1, const float* bih1, const float* bhh1,
            const float* Wq, const float* Wk, const float* Wv,
            __bf16* Wcat0, __bf16* Wcat1, float* bz0, float* bz1,
            __bf16* Wqb, __bf16* Wkb, __bf16* Wvb) {
  int stride = gridDim.x * blockDim.x;
  int tid0 = blockIdx.x * blockDim.x + threadIdx.x;
  for (int i = tid0; i < 2048 * 1024; i += stride) {
    int n = i >> 10, kk = i & 1023;
    float w0 = (kk < 512) ? Wih0[n * 512 + kk] : Whh0[n * 512 + kk - 512];
    float w1 = (kk < 512) ? Wih1[n * 512 + kk] : Whh1[n * 512 + kk - 512];
    Wcat0[i] = f2bf(w0);
    Wcat1[i] = f2bf(w1);
  }
  for (int i = tid0; i < 512 * 512; i += stride) {
    Wqb[i] = f2bf(Wq[i]);
    Wkb[i] = f2bf(Wk[i]);
    Wvb[i] = f2bf(Wv[i]);
  }
  for (int i = tid0; i < 2048; i += stride) {
    bz0[i] = bih0[i] + bhh0[i];
    bz1[i] = bih1[i] + bhh1[i];
  }
}

// ---------------------------------------------------------------------------
// Initial state: torch view(NL,B,H) of (noise@Wh.T+bh) [B, NL*H]
// ---------------------------------------------------------------------------
__global__ void __launch_bounds__(256)
init_state(const float* noise, const float* Wh, const float* bh,
           const float* Wc, const float* bc,
           float* c0f, float* c1f, __bf16* XH0, __bf16* XH1) {
  int idx = blockIdx.x * blockDim.x + threadIdx.x;     // l*B*H + b*H + h
  if (idx >= 2 * 512 * 512) return;
  int l = idx >> 18;
  int rem = idx & 262143;
  int b = rem >> 9, h = rem & 511;
  int sb = idx >> 10, sj = idx & 1023;                 // source row/col of [B,1024]
  const float* nr = noise + sb * 100;
  float sh = bh[sj], sc = bc[sj];
  for (int j = 0; j < 100; ++j) {
    float nv = nr[j];
    sh += nv * Wh[sj * 100 + j];
    sc += nv * Wc[sj * 100 + j];
  }
  if (l == 0) {
    XH0[b * 1024 + 512 + h] = f2bf(sh);   // h0
    XH0[b * 1024 + h] = f2bf(0.f);        // x0 = 0
    c0f[b * 512 + h] = sc;
  } else {
    XH1[b * 1024 + 512 + h] = f2bf(sh);   // h1
    c1f[b * 512 + h] = sc;
  }
}

// ---------------------------------------------------------------------------
// Fused LSTM step for one layer:
//   z = XHin @ Wcat^T + bz  (WMMA, all 4 gates for a 32-row x 64-h slice)
//   c' = sig(f)*c + sig(i)*tanh(g);  h' = sig(o)*tanh(c')
// A-tile (32 rows x 1024 K, contiguous 64KB) is async-DMA'd into LDS once per
// block; K-loop is double-buffered (unroll-2) so B loads for chunk k+32 are in
// flight while WMMAs consume chunk k, and fragment registers never collide
// (no WMMA->VALU WAR hazard NOPs).
// grid: (8 h-slices, 16 m-tiles), block 256 (8 waves).
// Wave w: m-tile = w>>2, gate = w&3, 64 gate-columns (4 WMMA n-tiles).
// ---------------------------------------------------------------------------
__global__ void __launch_bounds__(256)
lstm_step_fused(const __bf16* __restrict__ XHin, const __bf16* __restrict__ Wcat,
                const float* __restrict__ bz, float* __restrict__ c,
                __bf16* __restrict__ outA, __bf16* __restrict__ outB,
                float* __restrict__ h1f, int t, int layer) {
  __shared__ union {
    __bf16 a[32 * 1024];        // 64 KB A-tile during the K loop
    float  z[32][4][64];        // 32 KB z staging afterwards
  } sh;
  const int K = 1024;
  int lane = threadIdx.x & 31, wave = threadIdx.x >> 5;
  int tid = threadIdx.x;
  int mt = wave >> 2;
  int gate = wave & 3;
  int n0g = blockIdx.x * 64;              // h-column base
  int ncol0 = gate * 512 + n0g;           // column base in z-space

  // ---- async DMA the contiguous 64KB A-tile into LDS ----
  {
    unsigned abase = (unsigned)(unsigned long long)(const void*)sh.a; // LDS byte offset
    const char* gsrc = (const char*)(XHin + (size_t)blockIdx.y * 32 * 1024);
#pragma unroll
    for (int it = 0; it < 16; ++it) {
      unsigned off = (unsigned)(tid + it * 256) * 16u;   // 0 .. 65520
      async_copy_b128(abase + off, gsrc + off);
    }
  }
  wait_async0();
  __syncthreads();

  v8f acc[4];
#pragma unroll
  for (int j = 0; j < 4; ++j) acc[j] = vzero8();

  // LDS A-fragment lane base (row = mt*16 + lane&15 within the tile)
  const __bf16* a_lane = sh.a + (size_t)(mt * 16 + (lane & 15)) * 1024 +
                         ((lane >> 4) ? 8 : 0);
  // global B-fragment lane base rows
  const __bf16* b_lane[4];
#pragma unroll
  for (int j = 0; j < 4; ++j)
    b_lane[j] = Wcat + (size_t)(ncol0 + j * 16 + (lane & 15)) * 1024 +
                ((lane >> 4) ? 8 : 0);

  FragU a0, a1, b0[4], b1[4];
  // preload chunk 0
  a0.h[0] = *(const v8bf*)(a_lane);
  a0.h[1] = *(const v8bf*)(a_lane + 16);
#pragma unroll
  for (int j = 0; j < 4; ++j) {
    b0[j].h[0] = *(const v8bf*)(b_lane[j]);
    b0[j].h[1] = *(const v8bf*)(b_lane[j] + 16);
  }

  for (int k0 = 0; k0 < K; k0 += 64) {
    // stage chunk k0+32 into buffer 1 (always valid: K % 64 == 0)
    a1.h[0] = *(const v8bf*)(a_lane + k0 + 32);
    a1.h[1] = *(const v8bf*)(a_lane + k0 + 48);
#pragma unroll
    for (int j = 0; j < 4; ++j) {
      if (k0 + 128 < K)
        __builtin_prefetch(b_lane[j] + k0 + 128, 0, 3);
      b1[j].h[0] = *(const v8bf*)(b_lane[j] + k0 + 32);
      b1[j].h[1] = *(const v8bf*)(b_lane[j] + k0 + 48);
    }
    // compute chunk k0 from buffer 0
#pragma unroll
    for (int j = 0; j < 4; ++j)
      acc[j] = __builtin_amdgcn_wmma_f32_16x16x32_bf16(false, a0.v, false, b0[j].v,
                                                       (short)0, acc[j], false, false);
    // stage chunk k0+64 into buffer 0
    if (k0 + 64 < K) {
      a0.h[0] = *(const v8bf*)(a_lane + k0 + 64);
      a0.h[1] = *(const v8bf*)(a_lane + k0 + 80);
#pragma unroll
      for (int j = 0; j < 4; ++j) {
        b0[j].h[0] = *(const v8bf*)(b_lane[j] + k0 + 64);
        b0[j].h[1] = *(const v8bf*)(b_lane[j] + k0 + 80);
      }
    }
    // compute chunk k0+32 from buffer 1
#pragma unroll
    for (int j = 0; j < 4; ++j)
      acc[j] = __builtin_amdgcn_wmma_f32_16x16x32_bf16(false, a1.v, false, b1[j].v,
                                                       (short)0, acc[j], false, false);
  }
  __syncthreads();   // all waves done reading sh.a before reusing as sh.z

  // z (+bias) -> LDS
#pragma unroll
  for (int j = 0; j < 4; ++j) {
    int n_local = j * 16 + (lane & 15);
    float bv = bz[ncol0 + n_local];
#pragma unroll
    for (int r = 0; r < 8; ++r) {
      int ml = mt * 16 + r + ((lane >> 4) ? 8 : 0);
      sh.z[ml][gate][n_local] = acc[j][r] + bv;
    }
  }
  __syncthreads();

#pragma unroll
  for (int u = 0; u < 8; ++u) {
    int e = tid + u * 256;                // 0..2047
    int ml = e >> 6, hcol = e & 63;
    int b = blockIdx.y * 32 + ml;
    int h = n0g + hcol;
    float gi = sh.z[ml][0][hcol], gf = sh.z[ml][1][hcol];
    float gg = sh.z[ml][2][hcol], go = sh.z[ml][3][hcol];
    int cidx = b * 512 + h;
    float cn = sigm(gf) * c[cidx] + sigm(gi) * tanhf(gg);
    float hn = sigm(go) * tanhf(cn);
    c[cidx] = cn;
    __bf16 hb = f2bf(hn);
    outA[b * 1024 + 512 + h] = hb;
    if (layer == 0) {
      outB[b * 1024 + h] = hb;
    } else {
      outB[((size_t)b * 150 + t) * 512 + h] = hb;
      h1f[cidx] = hn;
    }
  }
}

// ---------------------------------------------------------------------------
// QKV GEMM, bf16 out: C[M,N] = A[M,K]*W[N,K]^T + bias.
// 32 M-rows per wave (2 A-frags share each B-frag).  grid: (N/512, M/32).
// mode 0: out[m*512+n];  mode 1: vT[b][n][s] (S=150, pad 160)
// ---------------------------------------------------------------------------
__global__ void __launch_bounds__(256)
wmma_gemm_bf16out(const __bf16* __restrict__ A, int lda,
                  const __bf16* __restrict__ W, int ldw,
                  const float* __restrict__ bias, __bf16* __restrict__ out,
                  int K, int mode) {
  int lane = threadIdx.x & 31, wave = threadIdx.x >> 5;
  int m0 = blockIdx.y * 32;
  int n0 = blockIdx.x * 512 + wave * 64;
  v8f acc[2][4];
#pragma unroll
  for (int mi = 0; mi < 2; ++mi)
#pragma unroll
    for (int j = 0; j < 4; ++j) acc[mi][j] = vzero8();

  for (int k0 = 0; k0 < K; k0 += 32) {
    v16bf a0 = load_frag_a(A, lda, m0, k0, lane);
    v16bf a1 = load_frag_a(A, lda, m0 + 16, k0, lane);
#pragma unroll
    for (int j = 0; j < 4; ++j) {
      if (k0 + 64 < K)
        __builtin_prefetch(W + (size_t)(n0 + j * 16 + (lane & 15)) * ldw + k0 + 64, 0, 3);
      v16bf b = load_frag_b(W, ldw, n0 + j * 16, k0, lane);
      acc[0][j] = __builtin_amdgcn_wmma_f32_16x16x32_bf16(false, a0, false, b,
                                                          (short)0, acc[0][j], false, false);
      acc[1][j] = __builtin_amdgcn_wmma_f32_16x16x32_bf16(false, a1, false, b,
                                                          (short)0, acc[1][j], false, false);
    }
  }
#pragma unroll
  for (int mi = 0; mi < 2; ++mi) {
#pragma unroll
    for (int j = 0; j < 4; ++j) {
      int n = n0 + j * 16 + (lane & 15);
      float bv = bias[n];
#pragma unroll
      for (int r = 0; r < 8; ++r) {
        int m = m0 + mi * 16 + r + ((lane >> 4) ? 8 : 0);
        float val = acc[mi][j][r] + bv;
        if (mode == 0) {
          out[(size_t)m * 512 + n] = f2bf(val);
        } else {
          int b = m / 150, s = m - b * 150;
          out[((size_t)b * 512 + n) * 160 + s] = f2bf(val);
        }
      }
    }
  }
}

// ---------------------------------------------------------------------------
// Vocab head per step: logits (V=4), gumbel-softmax, x_new = y @ Wfc
// one block per batch row; writes x_{t+1} into next XH0 buffer
// ---------------------------------------------------------------------------
__global__ void __launch_bounds__(256)
vocab_step(const float* __restrict__ h1f, const float* __restrict__ Wfc,
           const float* __restrict__ bfc, const float* __restrict__ g_loop,
           __bf16* __restrict__ XH0next, int t) {
  int b = blockIdx.x, tid = threadIdx.x;
  __shared__ float red[256];
  __shared__ float ysh[4];
  const float* hr = h1f + (size_t)b * 512;
  float hv0 = hr[tid], hv1 = hr[tid + 256];
  for (int v = 0; v < 4; ++v) {
    red[tid] = hv0 * Wfc[v * 512 + tid] + hv1 * Wfc[v * 512 + tid + 256];
    __syncthreads();
    for (int o = 128; o > 0; o >>= 1) {
      if (tid < o) red[tid] += red[tid + o];
      __syncthreads();
    }
    if (tid == 0)
      ysh[v] = red[0] + bfc[v] + g_loop[((size_t)t * 512 + b) * 4 + v];
    __syncthreads();
  }
  if (tid == 0) {
    float mx = fmaxf(fmaxf(ysh[0], ysh[1]), fmaxf(ysh[2], ysh[3]));
    float e0 = __expf(ysh[0] - mx), e1 = __expf(ysh[1] - mx);
    float e2 = __expf(ysh[2] - mx), e3 = __expf(ysh[3] - mx);
    float inv = 1.f / (e0 + e1 + e2 + e3);
    ysh[0] = e0 * inv; ysh[1] = e1 * inv; ysh[2] = e2 * inv; ysh[3] = e3 * inv;
  }
  __syncthreads();
  float y0 = ysh[0], y1 = ysh[1], y2 = ysh[2], y3 = ysh[3];
  float xa = y0 * Wfc[tid] + y1 * Wfc[512 + tid] + y2 * Wfc[1024 + tid] + y3 * Wfc[1536 + tid];
  int t2 = tid + 256;
  float xb = y0 * Wfc[t2] + y1 * Wfc[512 + t2] + y2 * Wfc[1024 + t2] + y3 * Wfc[1536 + t2];
  XH0next[b * 1024 + tid] = f2bf(xa);
  XH0next[b * 1024 + t2] = f2bf(xb);
}

// ---------------------------------------------------------------------------
// Attention: per (b, 16-row s-tile): scores = q k^T (WMMA), softmax, P x V (WMMA)
// ---------------------------------------------------------------------------
#define SPAD 160
__global__ void __launch_bounds__(256)
attn_kernel(const __bf16* __restrict__ q, const __bf16* __restrict__ k,
            const __bf16* __restrict__ vT, __bf16* __restrict__ att) {
  __shared__ float sc[16][SPAD];
  __shared__ __bf16 pm[16][SPAD];
  const int S = 150, H = 512;
  int b = blockIdx.y, s0 = blockIdx.x * 16;
  int lane = threadIdx.x & 31, wave = threadIdx.x >> 5;
  const __bf16* qb = q + (size_t)b * S * H;
  const __bf16* kb = k + (size_t)b * S * H;
  const __bf16* vb = vT + (size_t)b * H * SPAD;

  // phase 1: scores
  for (int tt = wave; tt < SPAD / 16; tt += 8) {
    v8f acc = vzero8();
    int mrow = s0 + (lane & 15); if (mrow >= S) mrow = S - 1;
    int nrow = tt * 16 + (lane & 15); if (nrow >= S) nrow = S - 1;
    for (int k0 = 0; k0 < H; k0 += 32) {
      FragU ua, ub;
      const __bf16* pa = qb + (size_t)mrow * H + k0 + ((lane >> 4) ? 8 : 0);
      ua.h[0] = *(const v8bf*)pa;
      ua.h[1] = *(const v8bf*)(pa + 16);
      const __bf16* pb = kb + (size_t)nrow * H + k0 + ((lane >> 4) ? 8 : 0);
      ub.h[0] = *(const v8bf*)pb;
      ub.h[1] = *(const v8bf*)(pb + 16);
      acc = __builtin_amdgcn_wmma_f32_16x16x32_bf16(false, ua.v, false, ub.v,
                                                    (short)0, acc, false, false);
    }
#pragma unroll
    for (int r = 0; r < 8; ++r) {
      int ml = r + ((lane >> 4) ? 8 : 0);
      int col = tt * 16 + (lane & 15);
      sc[ml][col] = (col < S) ? acc[r] : -1e30f;
    }
  }
  __syncthreads();

  // phase 2: row softmax (threads 0..15, one row each)
  if (threadIdx.x < 16) {
    int rr = threadIdx.x;
    float mx = -3.4e38f;
    for (int cCol = 0; cCol < SPAD; ++cCol) mx = fmaxf(mx, sc[rr][cCol]);
    float sum = 0.f;
    for (int cCol = 0; cCol < SPAD; ++cCol) {
      float e = __expf(sc[rr][cCol] - mx);
      sc[rr][cCol] = e;
      sum += e;
    }
    float inv = 1.f / sum;
    for (int cCol = 0; cCol < SPAD; ++cCol) pm[rr][cCol] = f2bf(sc[rr][cCol] * inv);
  }
  __syncthreads();

  // phase 3: out = P x V (K = SPAD)
#pragma unroll
  for (int j = 0; j < 4; ++j) {
    int n0 = wave * 64 + j * 16;
    v8f acc = vzero8();
    for (int kc = 0; kc < SPAD; kc += 32) {
      v16bf a;
      int ml = lane & 15;
      int kb0 = kc + ((lane >> 4) ? 8 : 0);
#pragma unroll
      for (int e = 0; e < 8; ++e) {
        a[e] = pm[ml][kb0 + e];
        a[8 + e] = pm[ml][kb0 + 16 + e];
      }
      FragU ub;
      int hrow = n0 + (lane & 15);
      const __bf16* pb = vb + (size_t)hrow * SPAD + kc + ((lane >> 4) ? 8 : 0);
      ub.h[0] = *(const v8bf*)pb;
      ub.h[1] = *(const v8bf*)(pb + 16);
      acc = __builtin_amdgcn_wmma_f32_16x16x32_bf16(false, a, false, ub.v,
                                                    (short)0, acc, false, false);
    }
#pragma unroll
    for (int r = 0; r < 8; ++r) {
      int ml = r + ((lane >> 4) ? 8 : 0);
      int s = s0 + ml;
      if (s < S) {
        int h = n0 + (lane & 15);
        att[((size_t)b * S + s) * H + h] = f2bf(acc[r]);
      }
    }
  }
}

// ---------------------------------------------------------------------------
// Final: residual + layernorm + logits(V=4) + gumbel-softmax -> d_out
// one block per (b,s)
// ---------------------------------------------------------------------------
__global__ void __launch_bounds__(256)
ln_out_kernel(const __bf16* __restrict__ att, const __bf16* __restrict__ hs,
              const float* __restrict__ gammap, const float* __restrict__ ln_g,
              const float* __restrict__ ln_b, const float* __restrict__ Wfc,
              const float* __restrict__ bfc, const float* __restrict__ g_final,
              float* __restrict__ out) {
  int bs = blockIdx.x, tid = threadIdx.x;
  __shared__ float red[256];
  __shared__ float st[8];
  float g = gammap[0];
  size_t base = (size_t)bs * 512;
  float y0 = g * bf2f(att[base + tid]) + bf2f(hs[base + tid]);
  float y1 = g * bf2f(att[base + tid + 256]) + bf2f(hs[base + tid + 256]);
  red[tid] = y0 + y1;
  __syncthreads();
  for (int o = 128; o > 0; o >>= 1) { if (tid < o) red[tid] += red[tid + o]; __syncthreads(); }
  if (tid == 0) st[0] = red[0] * (1.f / 512.f);
  __syncthreads();
  float mu = st[0];
  float d0 = y0 - mu, d1 = y1 - mu;
  red[tid] = d0 * d0 + d1 * d1;
  __syncthreads();
  for (int o = 128; o > 0; o >>= 1) { if (tid < o) red[tid] += red[tid + o]; __syncthreads(); }
  if (tid == 0) st[1] = rsqrtf(red[0] * (1.f / 512.f) + 1e-5f);
  __syncthreads();
  float rstd = st[1];
  float n0 = d0 * rstd * ln_g[tid] + ln_b[tid];
  float n1 = d1 * rstd * ln_g[tid + 256] + ln_b[tid + 256];
  for (int v = 0; v < 4; ++v) {
    red[tid] = n0 * Wfc[v * 512 + tid] + n1 * Wfc[v * 512 + tid + 256];
    __syncthreads();
    for (int o = 128; o > 0; o >>= 1) { if (tid < o) red[tid] += red[tid + o]; __syncthreads(); }
    if (tid == 0) st[2 + v] = red[0] + bfc[v] + g_final[(size_t)bs * 4 + v];
    __syncthreads();
  }
  if (tid == 0) {
    float l0 = st[2], l1 = st[3], l2 = st[4], l3 = st[5];
    float mx = fmaxf(fmaxf(l0, l1), fmaxf(l2, l3));
    float e0 = __expf(l0 - mx), e1 = __expf(l1 - mx);
    float e2 = __expf(l2 - mx), e3 = __expf(l3 - mx);
    float inv = 1.f / (e0 + e1 + e2 + e3);
    float* op = out + (size_t)bs * 4;
    op[0] = e0 * inv; op[1] = e1 * inv; op[2] = e2 * inv; op[3] = e3 * inv;
  }
}

// ---------------------------------------------------------------------------
// Host launcher
// ---------------------------------------------------------------------------
extern "C" void kernel_launch(void* const* d_in, const int* in_sizes, int n_in,
                              void* d_out, int out_size, void* d_ws, size_t ws_size,
                              hipStream_t stream) {
  const float* noise  = (const float*)d_in[0];
  const float* Wh     = (const float*)d_in[1];
  const float* bh     = (const float*)d_in[2];
  const float* Wc     = (const float*)d_in[3];
  const float* bc     = (const float*)d_in[4];
  const float* W_ih0  = (const float*)d_in[5];
  const float* W_hh0  = (const float*)d_in[6];
  const float* b_ih0  = (const float*)d_in[7];
  const float* b_hh0  = (const float*)d_in[8];
  const float* W_ih1  = (const float*)d_in[9];
  const float* W_hh1  = (const float*)d_in[10];
  const float* b_ih1  = (const float*)d_in[11];
  const float* b_hh1  = (const float*)d_in[12];
  const float* Wq     = (const float*)d_in[13];
  const float* bq     = (const float*)d_in[14];
  const float* Wk     = (const float*)d_in[15];
  const float* bk     = (const float*)d_in[16];
  const float* Wv     = (const float*)d_in[17];
  const float* bv     = (const float*)d_in[18];
  const float* gamma  = (const float*)d_in[19];
  const float* ln_g   = (const float*)d_in[20];
  const float* ln_b   = (const float*)d_in[21];
  const float* Wfc    = (const float*)d_in[22];
  const float* bfc    = (const float*)d_in[23];
  const float* g_loop = (const float*)d_in[24];
  const float* g_fin  = (const float*)d_in[25];
  float* out = (float*)d_out;

  char* ws = (char*)d_ws;
  size_t off = 0;
  auto alloc = [&](size_t bytes) -> void* {
    void* p = ws + off;
    off = (off + bytes + 255) & ~(size_t)255;
    return p;
  };
  __bf16* Wcat0 = (__bf16*)alloc((size_t)2048 * 1024 * 2);
  __bf16* Wcat1 = (__bf16*)alloc((size_t)2048 * 1024 * 2);
  float*  bz0   = (float*) alloc(2048 * 4);
  float*  bz1   = (float*) alloc(2048 * 4);
  __bf16* Wqb   = (__bf16*)alloc((size_t)512 * 512 * 2);
  __bf16* Wkb   = (__bf16*)alloc((size_t)512 * 512 * 2);
  __bf16* Wvb   = (__bf16*)alloc((size_t)512 * 512 * 2);
  __bf16* XH0buf[2], *XH1buf[2];
  XH0buf[0] = (__bf16*)alloc((size_t)512 * 1024 * 2);
  XH0buf[1] = (__bf16*)alloc((size_t)512 * 1024 * 2);
  XH1buf[0] = (__bf16*)alloc((size_t)512 * 1024 * 2);
  XH1buf[1] = (__bf16*)alloc((size_t)512 * 1024 * 2);
  float*  c0f   = (float*) alloc((size_t)512 * 512 * 4);
  float*  c1f   = (float*) alloc((size_t)512 * 512 * 4);
  float*  h1f   = (float*) alloc((size_t)512 * 512 * 4);
  __bf16* hs    = (__bf16*)alloc((size_t)512 * 150 * 512 * 2);
  __bf16* qb    = (__bf16*)alloc((size_t)512 * 150 * 512 * 2);  // reused as att
  __bf16* kb    = (__bf16*)alloc((size_t)512 * 150 * 512 * 2);
  __bf16* vT    = (__bf16*)alloc((size_t)512 * 512 * 160 * 2);
  (void)ws_size; (void)in_sizes; (void)n_in; (void)out_size;

  prep_kernel<<<2048, 256, 0, stream>>>(W_ih0, W_hh0, b_ih0, b_hh0,
                                        W_ih1, W_hh1, b_ih1, b_hh1,
                                        Wq, Wk, Wv,
                                        Wcat0, Wcat1, bz0, bz1, Wqb, Wkb, Wvb);
  init_state<<<2048, 256, 0, stream>>>(noise, Wh, bh, Wc, bc,
                                       c0f, c1f, XH0buf[0], XH1buf[0]);

  for (int t = 0; t < 150; ++t) {
    __bf16* in0  = XH0buf[t & 1];
    __bf16* out0 = XH0buf[(t + 1) & 1];
    __bf16* in1  = XH1buf[t & 1];
    __bf16* out1 = XH1buf[(t + 1) & 1];
    // layer 0: reads [x_t | h0_{t-1}]; writes h0_t -> out0[:,512:] and in1[:,0:512]
    lstm_step_fused<<<dim3(8, 16), 256, 0, stream>>>(in0, Wcat0, bz0, c0f,
                                                     out0, in1, nullptr, t, 0);
    // layer 1: reads [h0_t | h1_{t-1}]; writes h1_t -> out1[:,512:], hs, h1f
    lstm_step_fused<<<dim3(8, 16), 256, 0, stream>>>(in1, Wcat1, bz1, c1f,
                                                     out1, hs, h1f, t, 1);
    // vocab head: writes x_{t+1} -> out0[:,0:512]
    vocab_step<<<512, 256, 0, stream>>>(h1f, Wfc, bfc, g_loop, out0, t);
  }

  // q, k row-major; v transposed (vT[b][h][t])
  wmma_gemm_bf16out<<<dim3(1, 2400), 256, 0, stream>>>(hs, 512, Wqb, 512, bq, qb, 512, 0);
  wmma_gemm_bf16out<<<dim3(1, 2400), 256, 0, stream>>>(hs, 512, Wkb, 512, bk, kb, 512, 0);
  wmma_gemm_bf16out<<<dim3(1, 2400), 256, 0, stream>>>(hs, 512, Wvb, 512, bv, vT, 512, 1);

  attn_kernel<<<dim3(10, 512), 256, 0, stream>>>(qb, kb, vT, /*att=*/qb);

  ln_out_kernel<<<76800, 256, 0, stream>>>(qb, hs, gamma, ln_g, ln_b,
                                           Wfc, bfc, g_fin, out);
}
